// LLPKTMultiType_14963666059398
// MI455X (gfx1250) — matrix operand order repaired and checked by reading, not compiled, not run on hardware
//
#include <hip/hip_runtime.h>
#include <hip/hip_bf16.h>
#include <math.h>

typedef __attribute__((ext_vector_type(16))) _Float16 v16h;
typedef __attribute__((ext_vector_type(8)))  float    v8f;

#define DD 64
#define CC 50
#define SS 50
#define LL 16
#define BB 32
#define H4C 256

enum { F_PROJ = 1, F_CORR = 2, F_XALL = 4, F_XL15 = 8 };

__device__ __forceinline__ float fast_rcp(float x) { return __builtin_amdgcn_rcpf(x); }
__device__ __forceinline__ float fast_sigm(float x) {
    return fast_rcp(1.0f + __expf(-x));
}
__device__ __forceinline__ float fast_tanh(float x) {
    // tanh(x) = 1 - 2/(exp(2x)+1); saturates correctly at +-1
    return 1.0f - 2.0f * fast_rcp(__expf(2.0f * x) + 1.0f);
}

// ---------------------------------------------------------------------------
// Kernel 1: batched row ops via WMMA.
//   er = sigmoid(X*We + be), ad = tanh(X*Wa + ba)   (flags & F_PROJ)
//   cw = softmax_{c<50}(X * key^T)                   (flags & F_CORR)
// 128 threads = 4 waves, one 16-row tile per wave.
// Weights are staged in LDS pre-swizzled into the WMMA B-operand register
// layout (one v16h per (tile,lane)), so each operand is a single 32B LDS read.
// A fragments are loaded straight from global (float4) and packed in regs.
// NOTE: nrows must be a multiple of 64 (all call sites satisfy this).
// ---------------------------------------------------------------------------
__global__ __launch_bounds__(128) void rowop_wmma_kernel(
    const int* __restrict__ idx, const float* __restrict__ emb, int nrows,
    const float* __restrict__ We, const float* __restrict__ be,
    const float* __restrict__ Wa, const float* __restrict__ ba,
    const float* __restrict__ key,
    float* __restrict__ out_er, float* __restrict__ out_ad,
    float* __restrict__ out_cw, float* __restrict__ out_x,
    int flags)
{
    // B-operand layout: tile index tIdx = (k0/32)*4 + (n0/16), per-lane v16h.
    __shared__ v16h pkWe[8][32];
    __shared__ v16h pkWa[8][32];
    __shared__ v16h pkKt[8][32];
    __shared__ float sLog[4][16][DD];

    const int t    = threadIdx.x;
    const int lane = t & 31;
    const int wv   = t >> 5;

    // cooperative weight staging, directly into B-operand layout
    {
        _Float16* pe = (_Float16*)pkWe;
        _Float16* pa = (_Float16*)pkWa;
        _Float16* pk = (_Float16*)pkKt;
        for (int e = t; e < 8 * 32 * 16; e += 128) {
            int tIdx  = e >> 9;
            int lane2 = (e >> 4) & 31;
            int h     = e & 15;
            int K = ((tIdx >> 2) << 5) + ((h & 8) << 1) + ((lane2 >> 4) << 3) + (h & 7);   // k0 + 16*(h/8) + 8*half + h%8
            int N = ((tIdx & 3) << 4) + (lane2 & 15);               // n0 + nl
            if (flags & F_PROJ) {
                pe[e] = (_Float16)We[K * DD + N];
                pa[e] = (_Float16)Wa[K * DD + N];
            }
            if (flags & F_CORR)
                pk[e] = (N < CC) ? (_Float16)key[N * DD + K] : (_Float16)0.0f;
        }
    }
    __syncthreads();

    const int r0 = (blockIdx.x * 4 + wv) * 16;

    // ---- A fragments straight from global -------------------------------
    const int m   = lane & 15;
    const int kbA = (lane >> 4) ? 8 : 0;
    const float* xr = emb + (size_t)idx[r0 + m] * DD;
    float4 u0 = *(const float4*)(xr + kbA);
    float4 u1 = *(const float4*)(xr + kbA + 4);
    float4 u2 = *(const float4*)(xr + 16 + kbA);
    float4 u3 = *(const float4*)(xr + 16 + kbA + 4);
    float4 u4 = *(const float4*)(xr + 32 + kbA);
    float4 u5 = *(const float4*)(xr + 32 + kbA + 4);
    float4 u6 = *(const float4*)(xr + 48 + kbA);
    float4 u7 = *(const float4*)(xr + 48 + kbA + 4);

    if (flags & F_XALL) {
        float* xo = out_x + (size_t)(r0 + m) * DD;
        *(float4*)(xo + kbA)          = u0;  *(float4*)(xo + kbA + 4)      = u1;
        *(float4*)(xo + 16 + kbA)     = u2;  *(float4*)(xo + 16 + kbA + 4) = u3;
        *(float4*)(xo + 32 + kbA)     = u4;  *(float4*)(xo + 32 + kbA + 4) = u5;
        *(float4*)(xo + 48 + kbA)     = u6;  *(float4*)(xo + 48 + kbA + 4) = u7;
    }
    if ((flags & F_XL15) && m == 15) {
        float* xo = out_x + (size_t)(r0 >> 4) * DD;
        *(float4*)(xo + kbA)          = u0;  *(float4*)(xo + kbA + 4)      = u1;
        *(float4*)(xo + 16 + kbA)     = u2;  *(float4*)(xo + 16 + kbA + 4) = u3;
        *(float4*)(xo + 32 + kbA)     = u4;  *(float4*)(xo + 32 + kbA + 4) = u5;
        *(float4*)(xo + 48 + kbA)     = u6;  *(float4*)(xo + 48 + kbA + 4) = u7;
    }

    v16h A0, A1;
    A0[0]=(_Float16)u0.x; A0[1]=(_Float16)u0.y; A0[2]=(_Float16)u0.z; A0[3]=(_Float16)u0.w;
    A0[4]=(_Float16)u1.x; A0[5]=(_Float16)u1.y; A0[6]=(_Float16)u1.z; A0[7]=(_Float16)u1.w;
    A0[8]=(_Float16)u2.x; A0[9]=(_Float16)u2.y; A0[10]=(_Float16)u2.z; A0[11]=(_Float16)u2.w;
    A0[12]=(_Float16)u3.x; A0[13]=(_Float16)u3.y; A0[14]=(_Float16)u3.z; A0[15]=(_Float16)u3.w;
    A1[0]=(_Float16)u4.x; A1[1]=(_Float16)u4.y; A1[2]=(_Float16)u4.z; A1[3]=(_Float16)u4.w;
    A1[4]=(_Float16)u5.x; A1[5]=(_Float16)u5.y; A1[6]=(_Float16)u5.z; A1[7]=(_Float16)u5.w;
    A1[8]=(_Float16)u6.x; A1[9]=(_Float16)u6.y; A1[10]=(_Float16)u6.z; A1[11]=(_Float16)u6.w;
    A1[12]=(_Float16)u7.x; A1[13]=(_Float16)u7.y; A1[14]=(_Float16)u7.z; A1[15]=(_Float16)u7.w;

    const int nl = lane & 15;
    const int mb = (lane >> 4) * 8;

    if (flags & F_PROJ) {
        #pragma unroll
        for (int n0 = 0; n0 < DD; n0 += 16) {
            const int tn = n0 >> 4;
            v8f acc = {};
            acc = __builtin_amdgcn_wmma_f32_16x16x32_f16(false, A0, false, pkWe[tn][lane],     (short)0, acc, false, false);
            acc = __builtin_amdgcn_wmma_f32_16x16x32_f16(false, A1, false, pkWe[4 + tn][lane], (short)0, acc, false, false);
            float bias = be[n0 + nl];
            #pragma unroll
            for (int v = 0; v < 8; ++v)
                out_er[(size_t)(r0 + mb + v) * DD + n0 + nl] = fast_sigm(acc[v] + bias);

            v8f ac2 = {};
            ac2 = __builtin_amdgcn_wmma_f32_16x16x32_f16(false, A0, false, pkWa[tn][lane],     (short)0, ac2, false, false);
            ac2 = __builtin_amdgcn_wmma_f32_16x16x32_f16(false, A1, false, pkWa[4 + tn][lane], (short)0, ac2, false, false);
            float bias2 = ba[n0 + nl];
            #pragma unroll
            for (int v = 0; v < 8; ++v)
                out_ad[(size_t)(r0 + mb + v) * DD + n0 + nl] = fast_tanh(ac2[v] + bias2);
        }
    }

    if (flags & F_CORR) {
        #pragma unroll
        for (int n0 = 0; n0 < DD; n0 += 16) {
            const int tn = n0 >> 4;
            v8f acc = {};
            acc = __builtin_amdgcn_wmma_f32_16x16x32_f16(false, A0, false, pkKt[tn][lane],     (short)0, acc, false, false);
            acc = __builtin_amdgcn_wmma_f32_16x16x32_f16(false, A1, false, pkKt[4 + tn][lane], (short)0, acc, false, false);
            #pragma unroll
            for (int v = 0; v < 8; ++v)
                sLog[wv][mb + v][n0 + nl] = acc[v];
        }
        // per-row softmax over the first 50 logits; lanes 0..15 = rows
        if (lane < 16) {
            int r = r0 + lane;
            float mx = -1e30f;
            for (int c = 0; c < CC; ++c) mx = fmaxf(mx, sLog[wv][lane][c]);
            float sum = 0.0f;
            for (int c = 0; c < CC; ++c) {
                float e = __expf(sLog[wv][lane][c] - mx);
                sLog[wv][lane][c] = e;
                sum += e;
            }
            float inv = fast_rcp(sum);
            for (int c = 0; c < CC; ++c)
                out_cw[(size_t)r * CC + c] = sLog[wv][lane][c] * inv;
        }
    }
}

// ---------------------------------------------------------------------------
// Kernel 2: sequential scan. One block (256 thr, 8 waves) per batch element.
// State M (50x64 f32 = 12.8KB) lives in LDS for the whole sequence.
// ---------------------------------------------------------------------------
__global__ __launch_bounds__(256) void seq_scan_kernel(
    const float* __restrict__ M0,
    const float* __restrict__ q_e,  const float* __restrict__ q_cw,
    const float* __restrict__ q_er, const float* __restrict__ q_ad,
    const float* __restrict__ l_cw, const float* __restrict__ l_er,
    const float* __restrict__ l_ad, const float* __restrict__ le_last,
    const float* __restrict__ gam,  const float* __restrict__ bet,
    const float* __restrict__ W0,   const float* __restrict__ b0,
    const float* __restrict__ W1,   const float* __restrict__ b1,
    const float* __restrict__ Wout, const float* __restrict__ bout,
    float* __restrict__ out)
{
    __shared__ float sM[CC * DD];
    __shared__ float sw[CC];
    __shared__ float se[DD], sa[DD], sLr[DD];
    __shared__ float sx[H4C], sxn[H4C], sh1[H4C];
    __shared__ float sred[H4C], sred2[H4C];

    const int b = blockIdx.x, t = threadIdx.x;
    for (int i = t; i < CC * DD; i += 256) sM[i] = M0[i];
    __syncthreads();

    for (int s = 0; s < SS; ++s) {
        const size_t bs = (size_t)b * SS + s;
        // q_read with M at step entry
        if (t < DD) {
            float acc = 0.0f;
            for (int c = 0; c < CC; ++c) acc += q_cw[bs * CC + c] * sM[c * DD + t];
            sx[t] = acc;
            sLr[t] = 0.0f;
        }
        __syncthreads();

        for (int l = 0; l < LL; ++l) {
            const size_t r = bs * LL + l;
            if (t < CC) sw[t] = l_cw[r * CC + t];
            if (t < DD) { se[t] = l_er[r * DD + t]; sa[t] = l_ad[r * DD + t]; }
            __syncthreads();
            if (t < DD) {
                float acc = 0.0f;
                for (int c = 0; c < CC; ++c) acc += sw[c] * sM[c * DD + t];
                sLr[t] += acc;
            }
            __syncthreads();
            for (int i = t; i < CC * DD; i += 256) {
                int c = i >> 6, d = i & 63;
                float wc = sw[c];
                sM[i] = sM[i] * (1.0f - wc * se[d]) + wc * sa[d];
            }
            __syncthreads();
        }

        if (t < DD) {
            sx[DD + t]     = q_e[bs * DD + t];
            sx[2 * DD + t] = sLr[t];
            sx[3 * DD + t] = le_last[bs * DD + t];
        }
        __syncthreads();

        // layernorm over 256
        float xv = sx[t];
        sred[t] = xv; sred2[t] = xv * xv;
        __syncthreads();
        for (int off = 128; off > 0; off >>= 1) {
            if (t < off) { sred[t] += sred[t + off]; sred2[t] += sred2[t + off]; }
            __syncthreads();
        }
        float mu  = sred[0] * (1.0f / H4C);
        float var = sred2[0] * (1.0f / H4C) - mu * mu;
        float xn  = (xv - mu) * rsqrtf(var + 1e-5f) * gam[t] + bet[t];
        sxn[t] = xn;
        __builtin_prefetch(W1 + t * H4C, 0, 0);   // pull layer-1 weights early
        __syncthreads();

        float acc = b0[t];
        for (int i = 0; i < H4C; ++i) acc += sxn[i] * W0[i * H4C + t];
        sh1[t] = fmaxf(acc, 0.0f);
        __syncthreads();

        float acc2 = b1[t];
        for (int i = 0; i < H4C; ++i) acc2 += sh1[i] * W1[i * H4C + t];
        sred[t] = acc2 * Wout[t];
        __syncthreads();
        for (int off = 128; off > 0; off >>= 1) {
            if (t < off) sred[t] += sred[t + off];
            __syncthreads();
        }
        if (t == 0) out[bs] = fast_sigm(sred[0] + bout[0]);
        __syncthreads();

        // final write of M with (qcw, q_er, q_ad)
        if (t < CC) sw[t] = q_cw[bs * CC + t];
        if (t < DD) { se[t] = q_er[bs * DD + t]; sa[t] = q_ad[bs * DD + t]; }
        __syncthreads();
        for (int i = t; i < CC * DD; i += 256) {
            int c = i >> 6, d = i & 63;
            float wc = sw[c];
            sM[i] = sM[i] * (1.0f - wc * se[d]) + wc * sa[d];
        }
        __syncthreads();
    }
}

extern "C" void kernel_launch(void* const* d_in, const int* in_sizes, int n_in,
                              void* d_out, int out_size, void* d_ws, size_t ws_size,
                              hipStream_t stream) {
    const int*   q_data  = (const int*)d_in[0];
    const int*   qa_data = (const int*)d_in[1];
    const int*   l_data  = (const int*)d_in[2];
    // d_in[3] users, d_in[7] u_embed: unused by forward
    const float* key     = (const float*)d_in[4];
    const float* q_embed = (const float*)d_in[5];
    const float* qa_embed= (const float*)d_in[6];
    const float* M0      = (const float*)d_in[8];
    const float* W_e     = (const float*)d_in[9];
    const float* b_e     = (const float*)d_in[10];
    const float* W_a     = (const float*)d_in[11];
    const float* b_a     = (const float*)d_in[12];
    const float* ln_g    = (const float*)d_in[13];
    const float* ln_b    = (const float*)d_in[14];
    const float* W0      = (const float*)d_in[15];
    const float* b0      = (const float*)d_in[16];
    const float* W1      = (const float*)d_in[17];
    const float* b1      = (const float*)d_in[18];
    const float* W_out   = (const float*)d_in[19];
    const float* b_out   = (const float*)d_in[20];

    const int NL_ROWS = BB * SS * LL;   // 25600 (multiple of 64)
    const int NQ_ROWS = BB * SS;        // 1600  (multiple of 64)

    float* p = (float*)d_ws;
    float* ws_l_er = p;  p += (size_t)NL_ROWS * DD;
    float* ws_l_ad = p;  p += (size_t)NL_ROWS * DD;
    float* ws_l_cw = p;  p += (size_t)NL_ROWS * CC;
    float* ws_q_e  = p;  p += (size_t)NQ_ROWS * DD;
    float* ws_q_cw = p;  p += (size_t)NQ_ROWS * CC;
    float* ws_q_er = p;  p += (size_t)NQ_ROWS * DD;
    float* ws_q_ad = p;  p += (size_t)NQ_ROWS * DD;
    float* ws_le15 = p;  p += (size_t)NQ_ROWS * DD;

    // 1) l rows: proj + corr, and capture last-l embedding rows
    rowop_wmma_kernel<<<NL_ROWS / 64, 128, 0, stream>>>(
        l_data, q_embed, NL_ROWS, W_e, b_e, W_a, b_a, key,
        ws_l_er, ws_l_ad, ws_l_cw, ws_le15, F_PROJ | F_CORR | F_XL15);

    // 2) qa rows: proj only
    rowop_wmma_kernel<<<NQ_ROWS / 64, 128, 0, stream>>>(
        qa_data, qa_embed, NQ_ROWS, W_e, b_e, W_a, b_a, key,
        ws_q_er, ws_q_ad, nullptr, nullptr, F_PROJ);

    // 3) q rows: corr + store gathered embeddings
    rowop_wmma_kernel<<<NQ_ROWS / 64, 128, 0, stream>>>(
        q_data, q_embed, NQ_ROWS, W_e, b_e, W_a, b_a, key,
        nullptr, nullptr, ws_q_cw, ws_q_e, F_CORR | F_XALL);

    // 4) sequential scan, one WGP-resident state per batch element
    seq_scan_kernel<<<BB, 256, 0, stream>>>(
        M0, ws_q_e, ws_q_cw, ws_q_er, ws_q_ad,
        ws_l_cw, ws_l_er, ws_l_ad, ws_le15,
        ln_g, ln_b, W0, b0, W1, b1, W_out, b_out, (float*)d_out);
}
